// MultiHeadAttention_62251255988347
// MI455X (gfx1250) — compile-verified, matrix operand order checked by reading, not run on hardware
//
#include <hip/hip_runtime.h>

// ---------------------------------------------------------------------------
// MHA forward for MI455X (gfx1250): bf16 WMMA GEMMs + fused RoPE + flash attn
// B=2, S=2048, D=2048, H=16, hd=128
// GEMM: double-buffered async-to-LDS B tiles (128x64), 2 M-tiles/wave.
// Flash: double-buffered async-to-LDS K/V tiles shared by 4 waves per block.
// ---------------------------------------------------------------------------

typedef __attribute__((ext_vector_type(8)))  float   v8f;
typedef __attribute__((ext_vector_type(8)))  __bf16  v8bf;
typedef __attribute__((ext_vector_type(16))) __bf16  v16bf;

__device__ __forceinline__ v16bf cat16(v8bf lo, v8bf hi) {
    return __builtin_shufflevector(lo, hi, 0,1,2,3,4,5,6,7,8,9,10,11,12,13,14,15);
}

// float -> bf16 bits, round-to-nearest-even
__device__ __forceinline__ unsigned short f2bf(float f) {
    unsigned u = __float_as_uint(f);
    unsigned r = u + 0x7FFFu + ((u >> 16) & 1u);
    return (unsigned short)(r >> 16);
}

// async global->LDS copies (ASYNCcnt path); shared IOFFSET applies to both
// the LDS and global addresses per the ISA async pseudocode.
__device__ __forceinline__ void async_copy64(unsigned lds, const void* g) {
    asm volatile(
        "global_load_async_to_lds_b128 %0, %1, off\n\t"
        "global_load_async_to_lds_b128 %0, %1, off offset:16\n\t"
        "global_load_async_to_lds_b128 %0, %1, off offset:32\n\t"
        "global_load_async_to_lds_b128 %0, %1, off offset:48"
        :: "v"(lds), "v"(g) : "memory");
}
__device__ __forceinline__ void async_copy128(unsigned lds, const void* g) {
    asm volatile(
        "global_load_async_to_lds_b128 %0, %1, off\n\t"
        "global_load_async_to_lds_b128 %0, %1, off offset:16\n\t"
        "global_load_async_to_lds_b128 %0, %1, off offset:32\n\t"
        "global_load_async_to_lds_b128 %0, %1, off offset:48\n\t"
        "global_load_async_to_lds_b128 %0, %1, off offset:64\n\t"
        "global_load_async_to_lds_b128 %0, %1, off offset:80\n\t"
        "global_load_async_to_lds_b128 %0, %1, off offset:96\n\t"
        "global_load_async_to_lds_b128 %0, %1, off offset:112"
        :: "v"(lds), "v"(g) : "memory");
}
__device__ __forceinline__ void wait_async() {
    asm volatile("s_wait_asynccnt 0" ::: "memory");
}
__device__ __forceinline__ unsigned lds_off(const void* p) {
    // generic shared pointer: low 32 bits are the 0-based LDS offset
    return (unsigned)(unsigned long long)p;
}

// ---------------------------------------------------------------------------
// Elementwise f32 -> bf16 conversion (grid-stride)
// ---------------------------------------------------------------------------
__global__ void cvt_bf16_kernel(const float* __restrict__ src,
                                unsigned short* __restrict__ dst, long n) {
    long i = (long)blockIdx.x * blockDim.x + threadIdx.x;
    long stride = (long)gridDim.x * blockDim.x;
    for (; i < n; i += stride) dst[i] = f2bf(src[i]);
}

// ---------------------------------------------------------------------------
// GEMM: C[M,N] = A[M,K] * W[N,K]^T   (A, W bf16; accumulate f32)
// Block = 128 threads (4 waves), all on one head (128-col N tile).
// B tile (128 x 64 k) double-buffered in LDS; each wave computes two
// 16(M) x 128(N) tiles (B-fragment reused by 8 M-tiles per block).
// mode 0: RoPE epilogue (+qscale), write bf16 to [B,H,S,HD]
// mode 1: no RoPE, write bf16 TRANSPOSED to [B,H,HD,S]   (V)
// mode 2: no RoPE, write f32 to [B,S,D]                   (output proj)
// ---------------------------------------------------------------------------
__global__ void __launch_bounds__(128)
gemm_rope_kernel(const unsigned short* __restrict__ A,
                 const unsigned short* __restrict__ W,
                 unsigned short* __restrict__ outbf,
                 float* __restrict__ outf,
                 int M, int K, int S, int H, int HD,
                 int mode, float qscale)
{
    __shared__ __align__(16) unsigned short Bt[2][128 * 64]; // [buf][row][k]

    const int lane = threadIdx.x & 31;
    const int w    = threadIdx.x >> 5;
    const int h    = blockIdx.x % H;           // head / 128-col tile of W
    const int mb   = blockIdx.x / H;           // 128-row block of A
    const int mt0  = mb * 8 + w * 2;           // this wave's first 16-row tile

    const int n    = lane & 15;
    const int half = lane >> 4;

    v8f acc[2][8];
#pragma unroll
    for (int i = 0; i < 2; i++)
#pragma unroll
        for (int c = 0; c < 8; c++)
#pragma unroll
            for (int r = 0; r < 8; r++) acc[i][c][r] = 0.0f;

    // cooperative staging: thread tid owns row tid (128 bytes per k-stage)
    const int brow = threadIdx.x;
    const unsigned ldsB0 = lds_off(&Bt[0][brow * 64]);
    const unsigned ldsB1 = lds_off(&Bt[1][brow * 64]);
    const unsigned short* gB = W + (long)(h * HD + brow) * K;

    async_copy128(ldsB0, gB);                  // prologue: stage k-block 0
    int buf = 0;
    for (int k0 = 0; k0 < K; k0 += 64) {
        wait_async();
        __syncthreads();                       // tile `buf` visible; buf^1 free
        if (k0 + 64 < K)
            async_copy128(buf ? ldsB0 : ldsB1, gB + k0 + 64);   // prefetch

        const unsigned short* Bbase = &Bt[buf][0];
#pragma unroll
        for (int kcs = 0; kcs < 2; kcs++) {
            const int kc = k0 + kcs * 32;
            v16bf afr[2];
#pragma unroll
            for (int i = 0; i < 2; i++) {
                const unsigned short* pa =
                    A + (long)((mt0 + i) * 16 + n) * K + kc + 8 * half;
                afr[i] = cat16(*(const v8bf*)pa, *(const v8bf*)(pa + 16));
            }
#pragma unroll
            for (int c = 0; c < 8; c++) {
                const unsigned short* pb =
                    Bbase + (c * 16 + n) * 64 + kcs * 32 + 8 * half;
                v16bf bfr = cat16(*(const v8bf*)pb, *(const v8bf*)(pb + 16));
                acc[0][c] = __builtin_amdgcn_wmma_f32_16x16x32_bf16(
                    false, afr[0], false, bfr, (short)0, acc[0][c], false, false);
                acc[1][c] = __builtin_amdgcn_wmma_f32_16x16x32_bf16(
                    false, afr[1], false, bfr, (short)0, acc[1][c], false, false);
            }
        }
        buf ^= 1;
    }

    const int D_ = H * HD;
#pragma unroll
    for (int i = 0; i < 2; i++) {
        const int mt = mt0 + i;
        // ---- fused RoPE: pair col j (chunk c) with j+64 (chunk c+4) ----
        if (mode == 0) {
#pragma unroll
            for (int c = 0; c < 4; c++) {
#pragma unroll
                for (int r = 0; r < 8; r++) {
                    const int mrow = r + 8 * half;
                    const int srow = (int)(((long)mt * 16 + mrow) % S);
                    const int j = c * 16 + n;                          // 0..63
                    const float inv = __expf(-(float)j * 0.14391156831212787f);
                    const float th  = (float)srow * inv;
                    float sn, cs; __sincosf(th, &sn, &cs);
                    const float x1 = acc[i][c][r], x2 = acc[i][c + 4][r];
                    acc[i][c][r]     = (x1 * cs - x2 * sn) * qscale;
                    acc[i][c + 4][r] = (x2 * cs + x1 * sn) * qscale;
                }
            }
        }
#pragma unroll
        for (int r = 0; r < 8; r++) {
            const int mrow = r + 8 * half;
            const long grow = (long)mt * 16 + mrow;
            const int b    = (int)(grow / S);
            const int srow = (int)(grow % S);
#pragma unroll
            for (int c = 0; c < 8; c++) {
                const float v = acc[i][c][r];
                if (mode == 0) {
                    long idx = (((long)b * H + h) * S + srow) * HD + c * 16 + n;
                    outbf[idx] = f2bf(v);
                } else if (mode == 1) {
                    long idx = (((long)b * H + h) * HD + (c * 16 + n)) * (long)S + srow;
                    outbf[idx] = f2bf(v);
                } else {
                    long idx = grow * D_ + h * HD + c * 16 + n;
                    outf[idx] = v;
                }
            }
        }
    }
}

// ---------------------------------------------------------------------------
// Flash attention: block = 4 waves = 4 adjacent 16-row q-tiles of one (b,h).
// K/V k-tiles (32 x 128) double-buffered in LDS via async copies.
// Q/K: [B,H,S,HD] bf16 (RoPE + scale pre-applied). V: [B,H,HD,S] bf16.
// Writes context bf16 to [B,S,D].
// ---------------------------------------------------------------------------
__global__ void __launch_bounds__(128)
flash_attn_kernel(const unsigned short* __restrict__ Q,
                  const unsigned short* __restrict__ Kh,
                  const unsigned short* __restrict__ Vt,
                  unsigned short* __restrict__ ctx,
                  int B_, int H, int S, int HD)
{
    __shared__ __align__(16) unsigned short Kt[2][32 * 128]; // [buf][krow][d]
    __shared__ __align__(16) unsigned short Vl[2][128 * 32]; // [buf][col][krow]
    __shared__ __align__(16) unsigned short plds[4][16 * 32];

    const int lane = threadIdx.x & 31;
    const int w    = threadIdx.x >> 5;
    const int qblocks = S / 64;
    const int qb = blockIdx.x % qblocks;
    const int bh = blockIdx.x / qblocks;
    const int tbase = qb * 4;
    const int t  = tbase + w;                  // this wave's q-tile

    const int n    = lane & 15;
    const int half = lane >> 4;

    // Q A-fragments (16 x 128, resident)
    v16bf qf[4];
    const long qbase = ((long)bh * S + t * 16 + n) * HD;
#pragma unroll
    for (int kc = 0; kc < 4; kc++) {
        const unsigned short* p = Q + qbase + kc * 32 + 8 * half;
        qf[kc] = cat16(*(const v8bf*)p, *(const v8bf*)(p + 16));
    }

    v8f o[8];
    float rmax[8], rsum[8];
#pragma unroll
    for (int c = 0; c < 8; c++)
#pragma unroll
        for (int r = 0; r < 8; r++) o[c][r] = 0.0f;
#pragma unroll
    for (int r = 0; r < 8; r++) { rmax[r] = -1e30f; rsum[r] = 0.0f; }

    // staging ownership: K row-segment + one V column per thread
    const int krow = threadIdx.x >> 2;
    const int dseg = (threadIdx.x & 3) * 32;
    const unsigned ldsK0 = lds_off(&Kt[0][krow * 128 + dseg]);
    const unsigned ldsK1 = lds_off(&Kt[1][krow * 128 + dseg]);
    const unsigned ldsV0 = lds_off(&Vl[0][threadIdx.x * 32]);
    const unsigned ldsV1 = lds_off(&Vl[1][threadIdx.x * 32]);
    const unsigned short* gK = Kh + ((long)bh * S + krow) * HD + dseg;
    const unsigned short* gV = Vt + ((long)bh * HD + threadIdx.x) * (long)S;

    const int mylast = (t * 16 + 15) / 32;                 // causal bound (wave)
    const int ktmax  = ((tbase + 3) * 16 + 15) / 32;       // block bound

    async_copy64(ldsK0, gK);                               // prologue: kt = 0
    async_copy64(ldsV0, gV);
    int buf = 0;
    for (int kt = 0; kt <= ktmax; kt++) {
        wait_async();
        __syncthreads();                                   // tiles visible
        if (kt < ktmax) {                                  // prefetch kt+1
            async_copy64(buf ? ldsK0 : ldsK1, gK + (long)(kt + 1) * 32 * HD);
            async_copy64(buf ? ldsV0 : ldsV1, gV + (kt + 1) * 32);
        }

        if (kt <= mylast) {
            // ---- S(16x32) = Q * K^T ----
            v8f sc[2];
#pragma unroll
            for (int c = 0; c < 2; c++) {
#pragma unroll
                for (int r = 0; r < 8; r++) sc[c][r] = 0.0f;
#pragma unroll
                for (int kc = 0; kc < 4; kc++) {
                    const unsigned short* p =
                        &Kt[buf][(c * 16 + n) * 128 + kc * 32 + 8 * half];
                    v16bf bf_ = cat16(*(const v8bf*)p, *(const v8bf*)(p + 16));
                    sc[c] = __builtin_amdgcn_wmma_f32_16x16x32_bf16(
                        false, qf[kc], false, bf_, (short)0, sc[c], false, false);
                }
            }
            // ---- causal mask + online softmax ----
            float corr[8];
#pragma unroll
            for (int r = 0; r < 8; r++) {
                const int qg = t * 16 + r + 8 * half;
                float v0 = sc[0][r]; if (kt * 32 + n      > qg) v0 = -1e30f;
                float v1 = sc[1][r]; if (kt * 32 + 16 + n > qg) v1 = -1e30f;
                float lm = fmaxf(v0, v1);
                lm = fmaxf(lm, __shfl_xor(lm, 1, 16));
                lm = fmaxf(lm, __shfl_xor(lm, 2, 16));
                lm = fmaxf(lm, __shfl_xor(lm, 4, 16));
                lm = fmaxf(lm, __shfl_xor(lm, 8, 16));
                const float nm = fmaxf(rmax[r], lm);
                corr[r] = __expf(rmax[r] - nm);
                rmax[r] = nm;
                const float p0 = __expf(v0 - nm);
                const float p1 = __expf(v1 - nm);
                float rs = p0 + p1;
                rs += __shfl_xor(rs, 1, 16);
                rs += __shfl_xor(rs, 2, 16);
                rs += __shfl_xor(rs, 4, 16);
                rs += __shfl_xor(rs, 8, 16);
                rsum[r] = rsum[r] * corr[r] + rs;
                plds[w][(r + 8 * half) * 32 + n]      = f2bf(p0);
                plds[w][(r + 8 * half) * 32 + 16 + n] = f2bf(p1);
            }
#pragma unroll
            for (int c = 0; c < 8; c++)
#pragma unroll
                for (int r = 0; r < 8; r++) o[c][r] *= corr[r];

            asm volatile("s_wait_dscnt 0" ::: "memory");   // P staged (own wave)

            const unsigned short* pp = &plds[w][n * 32 + 8 * half];
            v16bf pf = cat16(*(const v8bf*)pp, *(const v8bf*)(pp + 16));

            // ---- O(16x128) += P(16x32) * V(32x128) ----
#pragma unroll
            for (int c = 0; c < 8; c++) {
                const unsigned short* pv = &Vl[buf][(c * 16 + n) * 32 + 8 * half];
                v16bf vf = cat16(*(const v8bf*)pv, *(const v8bf*)(pv + 16));
                o[c] = __builtin_amdgcn_wmma_f32_16x16x32_bf16(
                    false, pf, false, vf, (short)0, o[c], false, false);
            }
        }
        buf ^= 1;
    }

    // ---- normalize and write context [B,S,D] bf16 ----
    const int h = bh % H, b = bh / H;
    const int D_ = H * HD;
#pragma unroll
    for (int r = 0; r < 8; r++) {
        const float invl = 1.0f / rsum[r];
        const int srow = t * 16 + r + 8 * half;
#pragma unroll
        for (int c = 0; c < 8; c++) {
            long idx = ((long)b * S + srow) * D_ + h * HD + c * 16 + n;
            ctx[idx] = f2bf(o[c][r] * invl);
        }
    }
    (void)B_;
}

// ---------------------------------------------------------------------------
extern "C" void kernel_launch(void* const* d_in, const int* in_sizes, int n_in,
                              void* d_out, int out_size, void* d_ws, size_t ws_size,
                              hipStream_t stream) {
    (void)in_sizes; (void)n_in; (void)out_size; (void)ws_size;
    const int B = 2, S = 2048, D = 2048, H = 16, HD = 128;
    const float* x  = (const float*)d_in[0];
    const float* Wq = (const float*)d_in[1];
    const float* Wk = (const float*)d_in[2];
    const float* Wv = (const float*)d_in[3];
    const float* Wo = (const float*)d_in[4];
    float* out = (float*)d_out;

    const size_t nX = (size_t)B * S * D;   // 8.39M
    const size_t nW = (size_t)D * D;       // 4.19M
    unsigned short* xb  = (unsigned short*)d_ws;
    unsigned short* wqb = xb  + nX;
    unsigned short* wkb = wqb + nW;
    unsigned short* wvb = wkb + nW;
    unsigned short* wob = wvb + nW;
    unsigned short* qh  = wob + nW;        // [B,H,S,HD]
    unsigned short* kh  = qh  + nX;        // [B,H,S,HD]
    unsigned short* vt  = kh  + nX;        // [B,H,HD,S]
    unsigned short* ctx = vt  + nX;        // [B,S,D]

    cvt_bf16_kernel<<<1024, 256, 0, stream>>>(x,  xb,  (long)nX);
    cvt_bf16_kernel<<<1024, 256, 0, stream>>>(Wq, wqb, (long)nW);
    cvt_bf16_kernel<<<1024, 256, 0, stream>>>(Wk, wkb, (long)nW);
    cvt_bf16_kernel<<<1024, 256, 0, stream>>>(Wv, wvb, (long)nW);
    cvt_bf16_kernel<<<1024, 256, 0, stream>>>(Wo, wob, (long)nW);

    const int M = B * S;
    const int gemmBlocks = (M / 128) * H;      // 128-row blocks x 16 heads = 512
    const float qscale = 0.08838834764831845f; // 1/sqrt(128)

    gemm_rope_kernel<<<gemmBlocks, 128, 0, stream>>>(xb, wqb, qh, nullptr,
                                                     M, D, S, H, HD, 0, qscale);
    gemm_rope_kernel<<<gemmBlocks, 128, 0, stream>>>(xb, wkb, kh, nullptr,
                                                     M, D, S, H, HD, 0, 1.0f);
    gemm_rope_kernel<<<gemmBlocks, 128, 0, stream>>>(xb, wvb, vt, nullptr,
                                                     M, D, S, H, HD, 1, 1.0f);

    const int attnBlocks = B * H * (S / 64);   // 4 q-tiles per block = 1024
    flash_attn_kernel<<<attnBlocks, 128, 0, stream>>>(qh, kh, vt, ctx, B, H, S, HD);

    gemm_rope_kernel<<<gemmBlocks, 128, 0, stream>>>(ctx, wob, nullptr, out,
                                                     M, D, S, H, HD, 2, 1.0f);
}
